// MemoryLayerPlus_7670811590920
// MI455X (gfx1250) — matrix operand order repaired and checked by reading, not compiled, not run on hardware
//
#include <hip/hip_runtime.h>
#include <hip/hip_bf16.h>
#include <math.h>

typedef _Float16 v4h  __attribute__((ext_vector_type(4)));
typedef _Float16 v8h  __attribute__((ext_vector_type(8)));
typedef _Float16 v16h __attribute__((ext_vector_type(16)));
typedef float    v8f  __attribute__((ext_vector_type(8)));

#define HEAD 2
#define KDIM 128
#define KNN  16
#define KNUM 512
#define NEG_INF (-3.0e38f)

// ---------------------------------------------------------------------------
// f32 -> f16 conversion, 4 elements / thread
// ---------------------------------------------------------------------------
__global__ __launch_bounds__(256)
void cvt_f32_to_f16(const float* __restrict__ src, _Float16* __restrict__ dst, int n4)
{
    int i = blockIdx.x * 256 + threadIdx.x;
    if (i >= n4) return;
    float4 v = reinterpret_cast<const float4*>(src)[i];
    v4h h;
    h[0] = (_Float16)v.x; h[1] = (_Float16)v.y;
    h[2] = (_Float16)v.z; h[3] = (_Float16)v.w;
    reinterpret_cast<v4h*>(dst)[i] = h;
}

// ---------------------------------------------------------------------------
// Batched NT GEMM via v_wmma_f32_16x16x32_f16:
//   C[z][m][n] = sum_k A[z][m][k] * W[z][n][k]
// A: M x K f16 row-major (lda), W: N x K f16 row-major (ldw), C: f32 (ldc).
// One wave -> 16x64 tile (one A fragment reused across 4 B fragments).
// Block = 128 threads = 4 waves stacked along M -> 64x64 block tile.
// All dims are exact multiples of the tile here, so no bounds checks
// (M=4096; N in {512,1024}; K in {128,1024}).
// EPI: 0 = plain f32 store, 1 = silu
// ---------------------------------------------------------------------------
template<int EPI>
__global__ __launch_bounds__(128)
void wmma_gemm_nt(const _Float16* __restrict__ Ag, const _Float16* __restrict__ Wg,
                  float* __restrict__ Cg,
                  int K, int lda, int ldw, int ldc,
                  long sAz, long sWz, long sCz)
{
    const int lane = threadIdx.x & 31;
    const int wave = threadIdx.x >> 5;
    const _Float16* A = Ag + (long)blockIdx.z * sAz;
    const _Float16* W = Wg + (long)blockIdx.z * sWz;
    float*          C = Cg + (long)blockIdx.z * sCz;

    const int row0 = blockIdx.y * 64 + wave * 16;
    const int col0 = blockIdx.x * 64;

    // wave32 fragment layouts used here: A and B both take k = 16*(lane/16) + s
    const int mn  = lane & 15;               // M row (A) / N col (B, C)
    const int kba = (lane < 16) ? 0 : 16;    // A: K base (16 contiguous K values per lane)
    const int kbb = (lane < 16) ? 0 : 16;    // B: K base (16 contiguous K values per lane)

    const _Float16* aPtr  = A + (size_t)(row0 + mn) * lda + kba;
    const _Float16* wPtr0 = W + (size_t)(col0 +  0 + mn) * ldw + kbb;
    const _Float16* wPtr1 = W + (size_t)(col0 + 16 + mn) * ldw + kbb;
    const _Float16* wPtr2 = W + (size_t)(col0 + 32 + mn) * ldw + kbb;
    const _Float16* wPtr3 = W + (size_t)(col0 + 48 + mn) * ldw + kbb;

    v8f zero = {0,0,0,0,0,0,0,0};
    v8f acc0 = zero, acc1 = zero, acc2 = zero, acc3 = zero;

    for (int k0 = 0; k0 < K; k0 += 32) {
        v8h alo = *reinterpret_cast<const v8h*>(aPtr + k0);
        v8h ahi = *reinterpret_cast<const v8h*>(aPtr + k0 + 8);
        v16h a;
        #pragma unroll
        for (int i = 0; i < 8; ++i) { a[i] = alo[i]; a[8 + i] = ahi[i]; }
        v16h b0 = *reinterpret_cast<const v16h*>(wPtr0 + k0);
        v16h b1 = *reinterpret_cast<const v16h*>(wPtr1 + k0);
        v16h b2 = *reinterpret_cast<const v16h*>(wPtr2 + k0);
        v16h b3 = *reinterpret_cast<const v16h*>(wPtr3 + k0);
        acc0 = __builtin_amdgcn_wmma_f32_16x16x32_f16(false, a, false, b0, (short)0, acc0, false, false);
        acc1 = __builtin_amdgcn_wmma_f32_16x16x32_f16(false, a, false, b1, (short)0, acc1, false, false);
        acc2 = __builtin_amdgcn_wmma_f32_16x16x32_f16(false, a, false, b2, (short)0, acc2, false, false);
        acc3 = __builtin_amdgcn_wmma_f32_16x16x32_f16(false, a, false, b3, (short)0, acc3, false, false);
    }

    // C/D layout: VGPR r, lanes 0-15 -> M=r, N=lane; lanes 16-31 -> M=8+r, N=lane-16
    const int mb = (lane < 16) ? 0 : 8;
    float* cPtr = C + (size_t)(row0 + mb) * ldc + col0 + mn;
    #pragma unroll
    for (int r = 0; r < 8; ++r) {
        float v0 = acc0[r], v1 = acc1[r], v2 = acc2[r], v3 = acc3[r];
        if (EPI == 1) {
            v0 = v0 / (1.0f + __expf(-v0));
            v1 = v1 / (1.0f + __expf(-v1));
            v2 = v2 / (1.0f + __expf(-v2));
            v3 = v3 / (1.0f + __expf(-v3));
        }
        float* p = cPtr + (size_t)r * ldc;
        p[0] = v0; p[16] = v1; p[32] = v2; p[48] = v3;
    }
}

// ---------------------------------------------------------------------------
// Wave helpers
// ---------------------------------------------------------------------------
__device__ inline float waveSum(float x)
{
    #pragma unroll
    for (int off = 16; off >= 1; off >>= 1) x += __shfl_xor(x, off, 32);
    return x;
}

__device__ inline void waveArgMax(float& bv, int& bi)
{
    #pragma unroll
    for (int off = 16; off >= 1; off >>= 1) {
        float ov = __shfl_xor(bv, off, 32);
        int   oi = __shfl_xor(bi, off, 32);
        if (ov > bv || (ov == bv && oi < bi)) { bv = ov; bi = oi; }
    }
}

// ---------------------------------------------------------------------------
// LayerNorm over 128-wide rows of Q (fp32) -> Qn (f16), one wave per row
// ---------------------------------------------------------------------------
__global__ __launch_bounds__(256)
void qnorm_kernel(const float* __restrict__ Q, const float* __restrict__ g,
                  _Float16* __restrict__ Qn, int rows)
{
    const int wid  = (blockIdx.x * 256 + threadIdx.x) >> 5;
    const int lane = threadIdx.x & 31;
    if (wid >= rows) return;
    const float* q = Q + (size_t)wid * KDIM;
    float v0 = q[lane], v1 = q[lane + 32], v2 = q[lane + 64], v3 = q[lane + 96];
    float mean = waveSum(v0 + v1 + v2 + v3) * (1.0f / KDIM);
    v0 -= mean; v1 -= mean; v2 -= mean; v3 -= mean;
    float var = waveSum(v0*v0 + v1*v1 + v2*v2 + v3*v3) * (1.0f / KDIM);
    float inv = rsqrtf(var + 1e-5f);
    _Float16* o = Qn + (size_t)wid * KDIM;
    o[lane]      = (_Float16)(v0 * inv * g[lane]);
    o[lane + 32] = (_Float16)(v1 * inv * g[lane + 32]);
    o[lane + 64] = (_Float16)(v2 * inv * g[lane + 64]);
    o[lane + 96] = (_Float16)(v3 * inv * g[lane + 96]);
}

// ---------------------------------------------------------------------------
// Two top-16-of-512, top-16 of 256 pair sums, softmax. One wave per (t, h).
// Results are kept lane-distributed so register arrays are only ever
// statically indexed (no scratch).  Tie-break = lowest index (matches jax).
// scores layout: plane q = h*2+p, scores[(q*bs + t)*KNUM + n]
// ---------------------------------------------------------------------------
__global__ __launch_bounds__(256)
void topk_softmax_kernel(const float* __restrict__ scores,
                         int* __restrict__ outIdx, float* __restrict__ outProb,
                         int bs)
{
    const int wid  = (blockIdx.x * 256 + threadIdx.x) >> 5;
    const int lane = threadIdx.x & 31;
    const int t = wid >> 1;
    const int h = wid & 1;
    if (t >= bs) return;

    const float* s1 = scores + ((size_t)(h * 2 + 0) * bs + t) * KNUM;
    const float* s2 = scores + ((size_t)(h * 2 + 1) * bs + t) * KNUM;

    float v1[16], v2[16];
    #pragma unroll
    for (int i = 0; i < 16; ++i) { v1[i] = s1[lane + 32*i]; v2[i] = s2[lane + 32*i]; }

    // -- top-16 of s1: lane L keeps rank (L>>1)  (combo i1 = lane>>1) --
    float myT1 = 0.0f; int myI1 = 0;
    #pragma unroll
    for (int r = 0; r < 16; ++r) {
        float bv = NEG_INF; int bi = 0x7FFFFFFF;
        #pragma unroll
        for (int i = 0; i < 16; ++i) { int idx = lane + 32*i; if (v1[i] > bv) { bv = v1[i]; bi = idx; } }
        waveArgMax(bv, bi);
        if ((lane >> 1) == r) { myT1 = bv; myI1 = bi; }
        int slot = ((bi & 31) == lane) ? (bi >> 5) : -1;
        #pragma unroll
        for (int i = 0; i < 16; ++i) if (i == slot) v1[i] = NEG_INF;
    }

    // -- top-16 of s2: even lanes keep ranks 0..7, odd lanes ranks 8..15
    //    (combo i2 = 8*(lane&1) + j) --
    float c2v[8]; int c2i[8];
    #pragma unroll
    for (int j = 0; j < 8; ++j) { c2v[j] = NEG_INF; c2i[j] = 0; }
    #pragma unroll
    for (int r = 0; r < 16; ++r) {
        float bv = NEG_INF; int bi = 0x7FFFFFFF;
        #pragma unroll
        for (int i = 0; i < 16; ++i) { int idx = lane + 32*i; if (v2[i] > bv) { bv = v2[i]; bi = idx; } }
        waveArgMax(bv, bi);
        if ((r >> 3) == (lane & 1)) { c2v[r & 7] = bv; c2i[r & 7] = bi; }
        int slot = ((bi & 31) == lane) ? (bi >> 5) : -1;
        #pragma unroll
        for (int i = 0; i < 16; ++i) if (i == slot) v2[i] = NEG_INF;
    }

    // -- 256 combos, 8 per lane: flat c = lane*8 + j  (== i1*16 + i2) --
    float cv[8]; int cbig[8];
    #pragma unroll
    for (int j = 0; j < 8; ++j) { cv[j] = myT1 + c2v[j]; cbig[j] = myI1 * KNUM + c2i[j]; }

    float keepV = NEG_INF; int keepI = 0;
    #pragma unroll
    for (int r = 0; r < 16; ++r) {
        float bv = NEG_INF; int bc = 0x7FFFFFFF;
        #pragma unroll
        for (int j = 0; j < 8; ++j) { int c = lane * 8 + j; if (cv[j] > bv) { bv = cv[j]; bc = c; } }
        waveArgMax(bv, bc);
        const int ownerLane = bc >> 3, ownerSlot = bc & 7;
        int send = 0;
        #pragma unroll
        for (int j = 0; j < 8; ++j) if (ownerLane == lane && ownerSlot == j) send = cbig[j];
        int bigStar = __shfl(send, ownerLane, 32);
        if (lane == r) { keepV = bv; keepI = bigStar; }
        #pragma unroll
        for (int j = 0; j < 8; ++j) if (ownerLane == lane && ownerSlot == j) cv[j] = NEG_INF;
    }

    // -- softmax over lanes 0..15 (descending, max at lane 0) --
    const float m = __shfl(keepV, 0, 32);
    float e = (lane < 16) ? __expf(keepV - m) : 0.0f;
    const float s = waveSum(e);
    if (lane < 16) {
        const size_t base = ((size_t)t * HEAD + h) * KNN;
        outProb[base + lane] = e / s;
        outIdx[base + lane]  = keepI;
    }
}

// ---------------------------------------------------------------------------
// Weighted gather of 32 value rows + gate multiply -> f16 tmp.
// One block per token, 256 threads x float4 = 4KB row, coalesced.
// global_prefetch of the next gathered row overlaps the random-access latency.
// ---------------------------------------------------------------------------
__global__ __launch_bounds__(256)
void gather_gate_kernel(const float* __restrict__ table,
                        const int* __restrict__ idx, const float* __restrict__ prob,
                        const float* __restrict__ gate, _Float16* __restrict__ tmp,
                        int H)
{
    const int t   = blockIdx.x;
    const int tid = threadIdx.x;
    __shared__ int   sI[HEAD * KNN];
    __shared__ float sP[HEAD * KNN];
    if (tid < HEAD * KNN) {
        sI[tid] = idx [(size_t)t * HEAD * KNN + tid];
        sP[tid] = prob[(size_t)t * HEAD * KNN + tid];
    }
    __syncthreads();

    float ax = 0.f, ay = 0.f, az = 0.f, aw = 0.f;
    #pragma unroll 4
    for (int k = 0; k < HEAD * KNN; ++k) {
        if (k + 1 < HEAD * KNN)
            __builtin_prefetch(table + (size_t)sI[k + 1] * H + tid * 4, 0, 0);
        float4 v = reinterpret_cast<const float4*>(table + (size_t)sI[k] * H)[tid];
        float p = sP[k];
        ax += p * v.x; ay += p * v.y; az += p * v.z; aw += p * v.w;
    }
    float4 g = reinterpret_cast<const float4*>(gate + (size_t)t * H)[tid];
    v4h o;
    o[0] = (_Float16)(ax * g.x); o[1] = (_Float16)(ay * g.y);
    o[2] = (_Float16)(az * g.z); o[3] = (_Float16)(aw * g.w);
    reinterpret_cast<v4h*>(tmp + (size_t)t * H)[tid] = o;
}

// ---------------------------------------------------------------------------
extern "C" void kernel_launch(void* const* d_in, const int* in_sizes, int n_in,
                              void* d_out, int out_size, void* d_ws, size_t ws_size,
                              hipStream_t stream)
{
    (void)n_in; (void)out_size; (void)ws_size;
    const float* x     = (const float*)d_in[0];   // (bs, H)
    const float* wq    = (const float*)d_in[1];   // (512, H)
    const float* qnw   = (const float*)d_in[2];   // (128,)
    const float* keys  = (const float*)d_in[3];   // (2, 2, 512, 128)
    const float* table = (const float*)d_in[4];   // (262144, H)
    const float* wsw   = (const float*)d_in[5];   // (H, H)
    const float* wv    = (const float*)d_in[6];   // (H, H)
    float* out = (float*)d_out;

    const int H  = 1024;
    const int bs = in_sizes[0] / H;              // 4096
    const int QW = 2 * HEAD * KDIM;              // 512

    // workspace carve-out (256B aligned)
    char* ws = (char*)d_ws;
    size_t off = 0;
    auto alloc = [&](size_t bytes) -> void* {
        void* p = ws + off; off += (bytes + 255) & ~(size_t)255; return p;
    };
    _Float16* Xh   = (_Float16*)alloc((size_t)bs * H * 2);
    _Float16* Wqh  = (_Float16*)alloc((size_t)QW * H * 2);
    _Float16* Kh   = (_Float16*)alloc((size_t)HEAD * 2 * KNUM * KDIM * 2);
    _Float16* Wswh = (_Float16*)alloc((size_t)H * H * 2);
    _Float16* Wvh  = (_Float16*)alloc((size_t)H * H * 2);
    float*    Q    = (float*)   alloc((size_t)bs * QW * 4);
    _Float16* Qn   = (_Float16*)alloc((size_t)bs * QW * 2);
    float*    Sc   = (float*)   alloc((size_t)4 * bs * KNUM * 4);
    float*    Gate = (float*)   alloc((size_t)bs * H * 4);
    int*      Idx  = (int*)     alloc((size_t)bs * HEAD * KNN * 4);
    float*    Prob = (float*)   alloc((size_t)bs * HEAD * KNN * 4);
    _Float16* Tmp  = (_Float16*)alloc((size_t)bs * H * 2);

    auto cvt = [&](const float* s, _Float16* d, size_t n) {
        int n4 = (int)(n / 4);
        cvt_f32_to_f16<<<(n4 + 255) / 256, 256, 0, stream>>>(s, d, n4);
    };
    cvt(x,    Xh,   (size_t)bs * H);
    cvt(wq,   Wqh,  (size_t)QW * H);
    cvt(keys, Kh,   (size_t)HEAD * 2 * KNUM * KDIM);
    cvt(wsw,  Wswh, (size_t)H * H);
    cvt(wv,   Wvh,  (size_t)H * H);

    // Q = X @ Wq^T  (bs x 512, K=1024)
    wmma_gemm_nt<0><<<dim3(QW / 64, bs / 64, 1), 128, 0, stream>>>(
        Xh, Wqh, Q, H, H, H, QW, 0, 0, 0);

    // per-(head,half) layernorm of 128-wide rows, scaled by q_norm_w
    {
        int rows = bs * HEAD * 2;
        qnorm_kernel<<<(rows * 32 + 255) / 256, 256, 0, stream>>>(Q, qnw, Qn, rows);
    }

    // scores: 4 planes q=h*2+p: (bs x 512) = Qn[:, q*128:(q+1)*128] @ keys[h][p]^T
    wmma_gemm_nt<0><<<dim3(KNUM / 64, bs / 64, 4), 128, 0, stream>>>(
        Qn, Kh, Sc, KDIM, QW, KDIM, KNUM,
        (long)KDIM, (long)KNUM * KDIM, (long)bs * KNUM);

    // top-k + pair-sum top-k + softmax
    topk_softmax_kernel<<<(bs * HEAD * 32 + 255) / 256, 256, 0, stream>>>(
        Sc, Idx, Prob, bs);

    // gate = silu(X @ Wsw^T)
    wmma_gemm_nt<1><<<dim3(H / 64, bs / 64, 1), 128, 0, stream>>>(
        Xh, Wswh, Gate, H, H, H, H, 0, 0, 0);

    // tmp = (sum_k prob * table[idx]) * gate   -> f16
    gather_gate_kernel<<<bs, 256, 0, stream>>>(table, Idx, Prob, Gate, Tmp, H);

    // out = tmp @ Wv^T
    wmma_gemm_nt<0><<<dim3(H / 64, bs / 64, 1), 128, 0, stream>>>(
        Tmp, Wvh, out, H, H, H, H, 0, 0, 0);
}
